// MultiHeadAttention_16320875724873
// MI455X (gfx1250) — matrix operand order repaired and checked by reading, not compiled, not run on hardware
//
#include <hip/hip_runtime.h>

#define DEV __device__ __forceinline__

typedef __attribute__((ext_vector_type(16))) __bf16 v16bf;
typedef __attribute__((ext_vector_type(8)))  __bf16 v8bf;
typedef __attribute__((ext_vector_type(8)))  float  v8f;
typedef unsigned short u16;
typedef unsigned int   u32;

static constexpr int BATCH = 2;
static constexpr int SEQ   = 2048;
static constexpr int DIM   = 1024;
static constexpr int HEADS = 16;
static constexpr int HD    = 64;

// f32 -> bf16 (round to nearest even) for scalar / scattered stores
DEV u16 f2bf(float f) {
  u32 u = __builtin_bit_cast(u32, f);
  u32 r = (u + 0x7FFFu + ((u >> 16) & 1u)) >> 16;
  return (u16)r;
}

// packed 2xf32 -> 2xbf16 in one VALU op (v_cvt_pk_bf16_f32)
DEV u32 cvt_pk_bf16(float a, float b) {
  u32 d;
  asm("v_cvt_pk_bf16_f32 %0, %1, %2" : "=v"(d) : "v"(a), "v"(b));
  return d;
}

// per-lane async DMA: 16 bytes global -> LDS, tracked by ASYNCcnt
DEV void async_copy_b128(u32 lds_byte_addr, const void* gptr) {
  asm volatile("global_load_async_to_lds_b128 %0, %1, off"
               :: "v"(lds_byte_addr), "v"(gptr) : "memory");
}
DEV void wait_async0() { asm volatile("s_wait_asynccnt 0x0" ::: "memory"); }

// generic (flat) pointer to a __shared__ object -> 32-bit LDS byte address
DEV u32 lds_addr_of(const void* p) { return (u32)(size_t)p; }

DEV v8f wmma_bf16(v16bf a, v16bf b, v8f c) {
  // (neg_a, A, neg_b, B, c_mod, C, reuse_a, reuse_b)
  return __builtin_amdgcn_wmma_f32_16x16x32_bf16(false, a, false, b, (short)0, c,
                                                 false, false);
}

// A fragment: 16(M) x 32(K), K-major source rows, stride ld (elements).
// ISA 7.12.2: lane<16 -> K {0..7,16..23}; lane>=16 -> K {8..15,24..31}; M = lane%16.
DEV v16bf load_frag_a(const u16* base, int ld) {
  const int lane = threadIdx.x & 31;
  const u16* p = base + (size_t)(lane & 15) * ld + ((lane >> 4) << 3);
  v8bf lo = *reinterpret_cast<const v8bf*>(p);
  v8bf hi = *reinterpret_cast<const v8bf*>(p + 16);
  v16bf f;
#pragma unroll
  for (int i = 0; i < 8; ++i) { f[i] = lo[i]; f[i + 8] = hi[i]; }
  return f;
}

// B fragment: 32(K) x 16(N); source is B^T rows (row n, K contiguous, stride ld).
// lane<16 -> K {0..7,16..23}; lane>=16 -> K {8..15,24..31}; N = lane%16.
DEV v16bf load_frag_b(const u16* base, int ld) {
  const int lane = threadIdx.x & 31;
  const u16* p = base + (size_t)(lane & 15) * ld + ((lane >> 4) << 3);
  v8bf lo = *reinterpret_cast<const v8bf*>(p);
  v8bf hi = *reinterpret_cast<const v8bf*>(p + 16);
  v16bf f;
#pragma unroll
  for (int i = 0; i < 8; ++i) { f[i] = lo[i]; f[i + 8] = hi[i]; }
  return f;
}

// ---------------------------------------------------------------------------
// Kernel 1: fused QKV projection. grid = (M/128, D/128, 3). 256 thr (8 waves).
// y = x @ W^T : both x and W are K-major, so A and B tiles load identically.
// z=0 -> Q [B,H,N,64] bf16 ; z=1 -> K same ; z=2 -> V transposed [B,H,64,N].
// ---------------------------------------------------------------------------
__global__ __launch_bounds__(256) void mha_qkv(
    const float* __restrict__ x, const float* __restrict__ Wq,
    const float* __restrict__ Wk, const float* __restrict__ Wv,
    u16* __restrict__ Qb, u16* __restrict__ Kb, u16* __restrict__ Vtb) {
  constexpr int LDT = 40;  // 80B row stride: 16B aligned, bank-staggered
  __shared__ __align__(16) u16 As[128 * LDT];
  __shared__ __align__(16) u16 Bs[128 * LDT];

  const int z  = blockIdx.z;
  const float* W = (z == 0) ? Wq : (z == 1) ? Wk : Wv;
  const int m0 = blockIdx.x * 128;
  const int n0 = blockIdx.y * 128;
  const int tid  = threadIdx.x;
  const int w    = tid >> 5;
  const int lane = tid & 31;
  const int rm = (w >> 2) * 64;  // wave row base (2 waves in M)
  const int cn = (w & 3) * 32;   // wave col base (4 waves in N)

  const v8f vzero = {0.f, 0.f, 0.f, 0.f, 0.f, 0.f, 0.f, 0.f};
  v8f acc[4][2];
#pragma unroll
  for (int mt = 0; mt < 4; ++mt)
#pragma unroll
    for (int nt = 0; nt < 2; ++nt) acc[mt][nt] = vzero;

  for (int k0 = 0; k0 < DIM; k0 += 32) {
#pragma unroll
    for (int i = 0; i < 4; ++i) {  // 1024 float4 chunks / 256 threads
      const int l = tid + i * 256;
      const int row = l >> 3;
      const int kq  = (l & 7) << 2;
      const float* pa = x + (size_t)(m0 + row) * DIM + k0 + kq;
      const float4 fa = *reinterpret_cast<const float4*>(pa);
      u32* da = reinterpret_cast<u32*>(&As[row * LDT + kq]);
      da[0] = cvt_pk_bf16(fa.x, fa.y);
      da[1] = cvt_pk_bf16(fa.z, fa.w);
      const float* pb = W + (size_t)(n0 + row) * DIM + k0 + kq;
      const float4 fb = *reinterpret_cast<const float4*>(pb);
      u32* db = reinterpret_cast<u32*>(&Bs[row * LDT + kq]);
      db[0] = cvt_pk_bf16(fb.x, fb.y);
      db[1] = cvt_pk_bf16(fb.z, fb.w);
      if (k0 + 32 < DIM) {  // warm caches for next K-slab (global_prefetch_b8)
        __builtin_prefetch(pa + 32, 0, 1);
        __builtin_prefetch(pb + 32, 0, 1);
      }
    }
    __syncthreads();

    v16bf af[4];
#pragma unroll
    for (int mt = 0; mt < 4; ++mt)
      af[mt] = load_frag_a(&As[(rm + mt * 16) * LDT], LDT);
#pragma unroll
    for (int nt = 0; nt < 2; ++nt) {
      const v16bf bfg = load_frag_b(&Bs[(cn + nt * 16) * LDT], LDT);
#pragma unroll
      for (int mt = 0; mt < 4; ++mt)
        acc[mt][nt] = wmma_bf16(af[mt], bfg, acc[mt][nt]);
    }
    __syncthreads();
  }

  // C layout: row M = vgpr + 8*(lane>=16), col N = lane%16
  const int half = lane >> 4, nl = lane & 15;
  u16* dstQK = (z == 0) ? Qb : Kb;
#pragma unroll
  for (int mt = 0; mt < 4; ++mt)
#pragma unroll
    for (int nt = 0; nt < 2; ++nt)
#pragma unroll
      for (int i = 0; i < 8; ++i) {
        const int m = m0 + rm + mt * 16 + half * 8 + i;
        const int n = n0 + cn + nt * 16 + nl;
        const u16 v = f2bf(acc[mt][nt][i]);
        const int b = m >> 11, tok = m & (SEQ - 1);
        const int h = n >> 6,  d   = n & (HD - 1);
        if (z < 2)
          dstQK[(((size_t)(b * HEADS + h) * SEQ + tok) << 6) + d] = v;
        else
          Vtb[((size_t)(b * HEADS + h) * HD + d) * SEQ + tok] = v;
      }
}

// ---------------------------------------------------------------------------
// Kernel 2: causal flash attention. grid = (SEQ/128, B*H). 256 thr (8 waves).
// Each wave owns 16 query rows; loops over key chunks of 32.
// ---------------------------------------------------------------------------
__global__ __launch_bounds__(256) void mha_attn(
    const u16* __restrict__ Qb, const u16* __restrict__ Kb,
    const u16* __restrict__ Vtb, u16* __restrict__ Cx) {
  __shared__ __align__(16) u16 Pl[8][16 * 32];  // per-wave P staging (re-swizzle)

  const int tid  = threadIdx.x;
  const int w    = tid >> 5;
  const int lane = tid & 31;
  const int half = lane >> 4, nl = lane & 15;
  const int bh = blockIdx.y;
  const int b  = bh >> 4, h = bh & 15;
  const int qbase = blockIdx.x * 128 + w * 16;

  const u16* Qrow = Qb + ((size_t)bh * SEQ + qbase) * HD;
  const v16bf qf0 = load_frag_a(Qrow, HD);       // d 0..31
  const v16bf qf1 = load_frag_a(Qrow + 32, HD);  // d 32..63

  const v8f vzero = {0.f, 0.f, 0.f, 0.f, 0.f, 0.f, 0.f, 0.f};
  v8f acc[4] = {vzero, vzero, vzero, vzero};  // ctx 16 x 64
  float mrow[8], lrow[8];
#pragma unroll
  for (int i = 0; i < 8; ++i) { mrow[i] = -1e30f; lrow[i] = 0.f; }

  const int kend = qbase + 16;  // causal: keys <= last row of this strip
  for (int kt = 0; kt < kend; kt += 32) {
    const u16* K0 = Kb + ((size_t)bh * SEQ + kt) * HD;
    const v16bf kf00 = load_frag_b(K0, HD);             // toks kt..+15, d 0..31
    const v16bf kf01 = load_frag_b(K0 + 32, HD);        // toks kt..+15, d 32..63
    const v16bf kf10 = load_frag_b(K0 + 16 * HD, HD);   // toks kt+16..+31
    const v16bf kf11 = load_frag_b(K0 + 16 * HD + 32, HD);

    v8f s0 = vzero, s1 = vzero;
    s0 = wmma_bf16(qf0, kf00, s0);
    s0 = wmma_bf16(qf1, kf01, s0);
    s1 = wmma_bf16(qf0, kf10, s1);
    s1 = wmma_bf16(qf1, kf11, s1);

    float p0[8], p1[8];
#pragma unroll
    for (int i = 0; i < 8; ++i) {
      const int qrow = qbase + half * 8 + i;
      float a = s0[i] * 0.125f;  // 1/sqrt(64)
      float c = s1[i] * 0.125f;
      if (kt + nl > qrow)      a = -1e9f;
      if (kt + 16 + nl > qrow) c = -1e9f;
      float mx = fmaxf(a, c);
#pragma unroll
      for (int o = 1; o < 16; o <<= 1) mx = fmaxf(mx, __shfl_xor(mx, o, 32));
      const float mnew = fmaxf(mrow[i], mx);
      const float sc = __expf(mrow[i] - mnew);
      mrow[i] = mnew;
      const float e0 = __expf(a - mnew);
      const float e1 = __expf(c - mnew);
      float rs = e0 + e1;
#pragma unroll
      for (int o = 1; o < 16; o <<= 1) rs += __shfl_xor(rs, o, 32);
      lrow[i] = lrow[i] * sc + rs;
#pragma unroll
      for (int t = 0; t < 4; ++t) acc[t][i] *= sc;
      p0[i] = e0; p1[i] = e1;
    }

    // stage P (16x32) to this wave's LDS, reload in A-fragment layout
    u16* pw = &Pl[w][0];
#pragma unroll
    for (int i = 0; i < 8; ++i) {
      const int r = half * 8 + i;
      pw[r * 32 + nl]      = f2bf(p0[i]);
      pw[r * 32 + 16 + nl] = f2bf(p1[i]);
    }
    const v16bf pf = load_frag_a(pw, 32);

#pragma unroll
    for (int t = 0; t < 4; ++t) {
      const u16* Vr = Vtb + ((size_t)bh * HD + t * 16) * SEQ + kt;  // Vt rows = V cols
      const v16bf vf = load_frag_b(Vr, SEQ);
      acc[t] = wmma_bf16(pf, vf, acc[t]);
    }
  }

  // ctx[b, tok, h*64 + d] bf16, normalized by row sum
#pragma unroll
  for (int t = 0; t < 4; ++t)
#pragma unroll
    for (int i = 0; i < 8; ++i) {
      const int tok = qbase + half * 8 + i;
      const float v = acc[t][i] / lrow[i];
      Cx[(size_t)(b * SEQ + tok) * DIM + h * HD + t * 16 + nl] = f2bf(v);
    }
}

// ---------------------------------------------------------------------------
// Kernel 3: output projection  out = ctx(bf16) @ Wo^T -> f32. grid = (32, 8).
// A tile is already bf16 -> copy via GLOBAL_LOAD_ASYNC_TO_LDS_B128 (ASYNCcnt).
// ---------------------------------------------------------------------------
__global__ __launch_bounds__(256) void mha_out(
    const u16* __restrict__ Cx, const float* __restrict__ Wo,
    float* __restrict__ out) {
  constexpr int LDT = 40;
  __shared__ __align__(16) u16 As[128 * LDT];
  __shared__ __align__(16) u16 Bs[128 * LDT];

  const int m0 = blockIdx.x * 128;
  const int n0 = blockIdx.y * 128;
  const int tid  = threadIdx.x;
  const int w    = tid >> 5;
  const int lane = tid & 31;
  const int rm = (w >> 2) * 64;
  const int cn = (w & 3) * 32;

  const v8f vzero = {0.f, 0.f, 0.f, 0.f, 0.f, 0.f, 0.f, 0.f};
  v8f acc[4][2];
#pragma unroll
  for (int mt = 0; mt < 4; ++mt)
#pragma unroll
    for (int nt = 0; nt < 2; ++nt) acc[mt][nt] = vzero;

  for (int k0 = 0; k0 < DIM; k0 += 32) {
#pragma unroll
    for (int i = 0; i < 2; ++i) {  // A: 512 async 16B copies, no VGPR round-trip
      const int l = tid + i * 256;
      const int row = l >> 2;
      const int kq  = (l & 3) << 3;
      const u16* gp = Cx + (size_t)(m0 + row) * DIM + k0 + kq;
      async_copy_b128(lds_addr_of(&As[row * LDT + kq]), gp);
    }
#pragma unroll
    for (int i = 0; i < 4; ++i) {  // B: convert Wo f32 -> bf16 (v_cvt_pk_bf16_f32)
      const int l = tid + i * 256;
      const int row = l >> 3;
      const int kq  = (l & 7) << 2;
      const float* pb = Wo + (size_t)(n0 + row) * DIM + k0 + kq;
      const float4 fb = *reinterpret_cast<const float4*>(pb);
      u32* db = reinterpret_cast<u32*>(&Bs[row * LDT + kq]);
      db[0] = cvt_pk_bf16(fb.x, fb.y);
      db[1] = cvt_pk_bf16(fb.z, fb.w);
      if (k0 + 32 < DIM) __builtin_prefetch(pb + 32, 0, 1);
    }
    wait_async0();   // our wave's DMA into LDS is done
    __syncthreads(); // everyone's is done

    v16bf af[4];
#pragma unroll
    for (int mt = 0; mt < 4; ++mt)
      af[mt] = load_frag_a(&As[(rm + mt * 16) * LDT], LDT);
#pragma unroll
    for (int nt = 0; nt < 2; ++nt) {
      const v16bf bfg = load_frag_b(&Bs[(cn + nt * 16) * LDT], LDT);
#pragma unroll
      for (int mt = 0; mt < 4; ++mt)
        acc[mt][nt] = wmma_bf16(af[mt], bfg, acc[mt][nt]);
    }
    __syncthreads();
  }

  const int half = lane >> 4, nl = lane & 15;
#pragma unroll
  for (int mt = 0; mt < 4; ++mt)
#pragma unroll
    for (int nt = 0; nt < 2; ++nt)
#pragma unroll
      for (int i = 0; i < 8; ++i) {
        const int m = m0 + rm + mt * 16 + half * 8 + i;
        const int n = n0 + cn + nt * 16 + nl;
        out[(size_t)m * DIM + n] = acc[mt][nt][i];
      }
}

// ---------------------------------------------------------------------------
extern "C" void kernel_launch(void* const* d_in, const int* in_sizes, int n_in,
                              void* d_out, int out_size, void* d_ws, size_t ws_size,
                              hipStream_t stream) {
  const float* x  = (const float*)d_in[0];
  const float* Wq = (const float*)d_in[1];
  const float* Wk = (const float*)d_in[2];
  const float* Wv = (const float*)d_in[3];
  const float* Wo = (const float*)d_in[4];
  float* out = (float*)d_out;

  const size_t SZ = (size_t)BATCH * HEADS * SEQ * HD;  // 4,194,304 elems (8 MB bf16)
  u16* Qb = (u16*)d_ws;
  u16* Kb = Qb + SZ;
  u16* Vt = Kb + SZ;
  u16* Cx = Vt + SZ;  // total 32 MB of workspace

  mha_qkv<<<dim3(BATCH * SEQ / 128, DIM / 128, 3), 256, 0, stream>>>(
      x, Wq, Wk, Wv, Qb, Kb, Vt);
  mha_attn<<<dim3(SEQ / 128, BATCH * HEADS, 1), 256, 0, stream>>>(Qb, Kb, Vt, Cx);
  mha_out<<<dim3(BATCH * SEQ / 128, DIM / 128, 1), 256, 0, stream>>>(Cx, Wo, out);
}
